// GraphConvolutionCollectLayer_50525995270255
// MI455X (gfx1250) — matrix operand order repaired and checked by reading, not compiled, not run on hardware
//
#include <hip/hip_runtime.h>
#include <hip/hip_bf16.h>
#include <stdint.h>

// ---------------------------------------------------------------------------
// Types for gfx1250 WMMA (bf16 inputs, fp32 accumulate)
// ---------------------------------------------------------------------------
typedef __attribute__((ext_vector_type(16))) __bf16 v16bf;
typedef __attribute__((ext_vector_type(8)))  __bf16 v8bf;
typedef __attribute__((ext_vector_type(8)))  float  v8f;

union Frag16 { v16bf v; v8bf h[2]; };

#define NN    8192   // target nodes
#define MM    8192   // source nodes
#define EE    512    // embed dim
#define HH    4      // heads
#define DH    128    // head dim
#define KMAX  32     // max degree

static __device__ __forceinline__ unsigned short f2bf(float f) {
    unsigned int u = __float_as_uint(f);
    unsigned int r = u + 0x7FFFu + ((u >> 16) & 1u);   // round-to-nearest-even
    return (unsigned short)(r >> 16);
}

// ---------------------------------------------------------------------------
// Weight convert + transpose: Wt[n][k] = bf16(W[k][n]), 512x512
// ---------------------------------------------------------------------------
__global__ void cvt_wt_kernel(const float* __restrict__ W,
                              unsigned short* __restrict__ Wt) {
    int i = blockIdx.x * 256 + threadIdx.x;     // 0 .. 512*512-1
    int n = i >> 9;
    int k = i & 511;
    Wt[(size_t)n * EE + k] = f2bf(W[(size_t)k * EE + n]);
}

// ---------------------------------------------------------------------------
// Row-major fp32 -> bf16 (4 elements/thread)
// ---------------------------------------------------------------------------
__global__ void cvt_rows_kernel(const float* __restrict__ in,
                                unsigned short* __restrict__ out, int n4) {
    int i = blockIdx.x * 256 + threadIdx.x;
    if (i < n4) {
        float4 v = ((const float4*)in)[i];
        unsigned long long p =
            (unsigned long long)f2bf(v.x) |
            ((unsigned long long)f2bf(v.y) << 16) |
            ((unsigned long long)f2bf(v.z) << 32) |
            ((unsigned long long)f2bf(v.w) << 48);
        ((unsigned long long*)out)[i] = p;
    }
}

// ---------------------------------------------------------------------------
// Build compact neighbor lists from dense 0/1 adjacency row (order-free:
// softmax attention is permutation invariant over neighbors).
// One 256-thread block per row, float4 coalesced scan of 8192 columns.
// ---------------------------------------------------------------------------
__global__ void build_adj_kernel(const float* __restrict__ adj,
                                 int* __restrict__ nbr_idx,
                                 int* __restrict__ nbr_deg) {
    int n = blockIdx.x;
    __shared__ int cnt;
    if (threadIdx.x == 0) cnt = 0;
    __syncthreads();
    const float4* row = (const float4*)(adj + (size_t)n * MM);
    for (int c4 = threadIdx.x; c4 < MM / 4; c4 += 256) {
        float4 v = row[c4];
        int base = c4 * 4;
        if (v.x > 0.5f) { int p = atomicAdd(&cnt, 1); if (p < KMAX) nbr_idx[n * KMAX + p] = base + 0; }
        if (v.y > 0.5f) { int p = atomicAdd(&cnt, 1); if (p < KMAX) nbr_idx[n * KMAX + p] = base + 1; }
        if (v.z > 0.5f) { int p = atomicAdd(&cnt, 1); if (p < KMAX) nbr_idx[n * KMAX + p] = base + 2; }
        if (v.w > 0.5f) { int p = atomicAdd(&cnt, 1); if (p < KMAX) nbr_idx[n * KMAX + p] = base + 3; }
    }
    __syncthreads();
    if (threadIdx.x == 0) nbr_deg[n] = (cnt > KMAX) ? KMAX : cnt;
}

// ---------------------------------------------------------------------------
// WMMA GEMM: C[rows,512] = act(A[rows,512](bf16) x Wt^T + bias)
//   Wt is pre-transposed bf16 [512 cols][512 k] so B-fragment loads are
//   contiguous. One wave computes a 16x64 output tile (A-frag reused x4).
//   Fragment layouts follow CDNA5 ISA §7.12.2 (wave32):
//     A: lane half h, row = lane%16, e<8 -> K=8h+e, e>=8 -> K=16+8h+(e-8)
//     B: col = lane%16, e<8 -> K=8h+e, e>=8 -> K=16+8h+(e-8)
//     D: m = j + 8h, n = lane%16
// ---------------------------------------------------------------------------
__global__ void gemm_bf16_kernel(const unsigned short* __restrict__ A,
                                 const unsigned short* __restrict__ Wt,
                                 const float* __restrict__ bias,
                                 float* __restrict__ outF,
                                 unsigned short* __restrict__ outB,
                                 const int* __restrict__ degMask,
                                 int relu) {
    int wave  = blockIdx.x * (blockDim.x >> 5) + (threadIdx.x >> 5);
    int lane  = threadIdx.x & 31;
    int tileM = wave >> 3;        // 512 row-tiles
    int tileN = wave & 7;         // 8 col-groups of 64
    int r     = lane & 15;
    int half  = lane >> 4;

    const unsigned short* Arow = A + (size_t)(tileM * 16 + r) * EE;

    v8f acc[4];
#pragma unroll
    for (int t = 0; t < 4; ++t) acc[t] = (v8f){0.f,0.f,0.f,0.f,0.f,0.f,0.f,0.f};

    for (int kb = 0; kb < EE; kb += 32) {
        Frag16 a;
        a.h[0] = *(const v8bf*)(Arow + kb + 8 * half);
        a.h[1] = *(const v8bf*)(Arow + kb + 16 + 8 * half);
#pragma unroll
        for (int t = 0; t < 4; ++t) {
            int col = tileN * 64 + t * 16 + r;
            const unsigned short* Brow = Wt + (size_t)col * EE + kb + 8 * half;
            Frag16 b;
            b.h[0] = ((const v8bf*)Brow)[0];
            b.h[1] = ((const v8bf*)Brow)[2];
            acc[t] = __builtin_amdgcn_wmma_f32_16x16x32_bf16(
                false, a.v, false, b.v, (short)0, acc[t], false, false);
        }
    }

    // epilogue: bias (+relu) (+deg>0 mask), write f32 and/or bf16
    int active[8];
#pragma unroll
    for (int j = 0; j < 8; ++j) {
        int row = tileM * 16 + j + 8 * half;
        active[j] = degMask ? (degMask[row] > 0) : 1;
    }
#pragma unroll
    for (int t = 0; t < 4; ++t) {
        int col = tileN * 64 + t * 16 + r;
        float bv = bias[col];
#pragma unroll
        for (int j = 0; j < 8; ++j) {
            int row = tileM * 16 + j + 8 * half;
            float v = acc[t][j] + bv;
            if (relu) v = fmaxf(v, 0.f);
            if (!active[j]) v = 0.f;
            if (outF) outF[(size_t)row * EE + col] = v;
            if (outB) outB[(size_t)row * EE + col] = f2bf(v);
        }
    }
}

// ---------------------------------------------------------------------------
// Attention: one block (4 waves) per node; wave h handles head h,
// lane = neighbor slot for scores / d-slice for context.
// ---------------------------------------------------------------------------
__global__ void attn_kernel(const float* __restrict__ q,
                            const float* __restrict__ Kall,
                            const float* __restrict__ Vall,
                            const int* __restrict__ nbr_idx,
                            const int* __restrict__ nbr_deg,
                            unsigned short* __restrict__ ctx_bf) {
    int n = blockIdx.x;
    __shared__ __align__(16) float qs[EE];
    int tid = threadIdx.x;                       // 128 threads
    ((float4*)qs)[tid] = ((const float4*)(q + (size_t)n * EE))[tid];
    __syncthreads();

    int h = tid >> 5, lane = tid & 31;
    int d = nbr_deg[n];
    if (d > KMAX) d = KMAX;
    int valid = lane < d;
    int row = valid ? nbr_idx[n * KMAX + lane] : 0;

    const float scale = 0.08838834764831845f;    // 1/sqrt(128)
    float s = -1e9f;
    if (valid) {
        const float4* kr = (const float4*)(Kall + (size_t)row * EE + h * DH);
        const float4* qh = (const float4*)(qs + h * DH);
        float acc = 0.f;
#pragma unroll
        for (int i = 0; i < DH / 4; ++i) {
            float4 kv = kr[i], qv = qh[i];
            acc += kv.x * qv.x + kv.y * qv.y + kv.z * qv.z + kv.w * qv.w;
        }
        s = acc * scale;
    }

    // wave32 softmax over neighbor slots
    float m = s;
#pragma unroll
    for (int off = 16; off; off >>= 1) m = fmaxf(m, __shfl_xor(m, off, 32));
    float e = __expf(s - m);
    float sum = e;
#pragma unroll
    for (int off = 16; off; off >>= 1) sum += __shfl_xor(sum, off, 32);
    float a = e / sum;

    // context: lane covers 4 contiguous dims of this head
    float4 acc4 = {0.f, 0.f, 0.f, 0.f};
    for (int j = 0; j < d; ++j) {
        float aj = __shfl(a, j, 32);
        int   rj = __shfl(row, j, 32);
        float4 vv = *(const float4*)(Vall + (size_t)rj * EE + h * DH + lane * 4);
        acc4.x += aj * vv.x; acc4.y += aj * vv.y;
        acc4.z += aj * vv.z; acc4.w += aj * vv.w;
    }
    unsigned long long p =
        (unsigned long long)f2bf(acc4.x) |
        ((unsigned long long)f2bf(acc4.y) << 16) |
        ((unsigned long long)f2bf(acc4.z) << 32) |
        ((unsigned long long)f2bf(acc4.w) << 48);
    *(unsigned long long*)(ctx_bf + (size_t)n * EE + h * DH + lane * 4) = p;
}

// ---------------------------------------------------------------------------
// Host-side launcher
// ---------------------------------------------------------------------------
extern "C" void kernel_launch(void* const* d_in, const int* in_sizes, int n_in,
                              void* d_out, int out_size, void* d_ws, size_t ws_size,
                              hipStream_t stream) {
    const float* target = (const float*)d_in[0];
    const float* source = (const float*)d_in[1];
    const float* adj    = (const float*)d_in[2];
    const float* Wf = (const float*)d_in[3];  const float* bf = (const float*)d_in[4];
    const float* Wq = (const float*)d_in[5];  const float* bq = (const float*)d_in[6];
    const float* Wk = (const float*)d_in[7];  const float* bk = (const float*)d_in[8];
    const float* Wv = (const float*)d_in[9];  const float* bv = (const float*)d_in[10];
    const float* Wo = (const float*)d_in[11]; const float* bo = (const float*)d_in[12];
    float* out = (float*)d_out;

    // workspace layout (all offsets multiple of 4096)
    char* ws = (char*)d_ws;
    size_t off = 0;
    const size_t WT_SZ  = (size_t)EE * EE * 2;        // 512 KB each
    const size_t BF_SZ  = (size_t)NN * EE * 2;        // 8 MB
    const size_t F32_SZ = (size_t)NN * EE * 4;        // 16 MB
    unsigned short* Wf_t = (unsigned short*)(ws + off); off += WT_SZ;
    unsigned short* Wq_t = (unsigned short*)(ws + off); off += WT_SZ;
    unsigned short* Wk_t = (unsigned short*)(ws + off); off += WT_SZ;
    unsigned short* Wv_t = (unsigned short*)(ws + off); off += WT_SZ;
    unsigned short* Wo_t = (unsigned short*)(ws + off); off += WT_SZ;
    unsigned short* tgt_bf = (unsigned short*)(ws + off); off += BF_SZ;
    unsigned short* src_bf = (unsigned short*)(ws + off); off += BF_SZ;
    unsigned short* s_bf   = (unsigned short*)(ws + off); off += BF_SZ;
    unsigned short* ctx_bf = (unsigned short*)(ws + off); off += BF_SZ;
    float* q_f = (float*)(ws + off); off += F32_SZ;
    float* K_f = (float*)(ws + off); off += F32_SZ;
    float* V_f = (float*)(ws + off); off += F32_SZ;
    int* nbr_idx = (int*)(ws + off); off += (size_t)NN * KMAX * 4;
    int* nbr_deg = (int*)(ws + off); off += (size_t)NN * 4;
    if (ws_size < off) return;

    // 1) weights: transpose + bf16
    cvt_wt_kernel<<<1024, 256, 0, stream>>>(Wf, Wf_t);
    cvt_wt_kernel<<<1024, 256, 0, stream>>>(Wq, Wq_t);
    cvt_wt_kernel<<<1024, 256, 0, stream>>>(Wk, Wk_t);
    cvt_wt_kernel<<<1024, 256, 0, stream>>>(Wv, Wv_t);
    cvt_wt_kernel<<<1024, 256, 0, stream>>>(Wo, Wo_t);

    // 2) activations to bf16
    int n4 = NN * EE / 4;
    cvt_rows_kernel<<<(n4 + 255) / 256, 256, 0, stream>>>(target, tgt_bf, n4);
    cvt_rows_kernel<<<(n4 + 255) / 256, 256, 0, stream>>>(source, src_bf, n4);

    // 3) neighbor lists from dense adjacency (256 MB scan — the roofline floor)
    build_adj_kernel<<<NN, 256, 0, stream>>>(adj, nbr_idx, nbr_deg);

    // 4-7) the five restructured GEMMs (rows=8192 -> 512 blocks of 8 waves)
    dim3 gblk(256), ggrd(NN / 16);
    // s = relu(source @ Wf + bf)  -> bf16 only
    gemm_bf16_kernel<<<ggrd, gblk, 0, stream>>>(src_bf, Wf_t, bf, nullptr, s_bf, nullptr, 1);
    // q = target @ Wq + bq        -> fp32
    gemm_bf16_kernel<<<ggrd, gblk, 0, stream>>>(tgt_bf, Wq_t, bq, q_f, nullptr, nullptr, 0);
    // K_all = s @ Wk + bk         -> fp32  (gather commutes with matmul)
    gemm_bf16_kernel<<<ggrd, gblk, 0, stream>>>(s_bf, Wk_t, bk, K_f, nullptr, nullptr, 0);
    // V_all = s @ Wv + bv         -> fp32
    gemm_bf16_kernel<<<ggrd, gblk, 0, stream>>>(s_bf, Wv_t, bv, V_f, nullptr, nullptr, 0);

    // 8) masked multi-head attention over gathered neighbors
    attn_kernel<<<NN, 128, 0, stream>>>(q_f, K_f, V_f, nbr_idx, nbr_deg, ctx_bf);

    // 9) out = (deg>0) ? ctx @ Wo + bo : 0   -> fp32 d_out
    gemm_bf16_kernel<<<ggrd, gblk, 0, stream>>>(ctx_bf, Wo_t, bo, out, nullptr, nbr_deg, 0);
}
